// PointPillars_87016037417630
// MI455X (gfx1250) — matrix operand order repaired and checked by reading, not compiled, not run on hardware
//
#include <hip/hip_runtime.h>
#include <hip/hip_bf16.h>

typedef __attribute__((ext_vector_type(16))) _Float16 v16h;
typedef __attribute__((ext_vector_type(8)))  float    v8f;

#define PNUM   48000
#define NPTSC  10
#define NXC    432
#define NYC    496
#define BATCH  4
#define H1     108
#define W1     124
#define HWF    (H1*W1)          // 13392
#define EPSN   1e-5f
#define VXC    0.16f
#define VYC    0.16f
#define XOFFC  0.08f            // VX/2 + XOFF
#define YOFFC  (-39.6f)         // VY/2 + YOFF

// ---------------------------------------------------------------------------
// generic elementwise
// ---------------------------------------------------------------------------
__global__ void zero_f32(float* p, long n) {
    long i = (long)blockIdx.x * blockDim.x + threadIdx.x;
    if (i < n) p[i] = 0.f;
}

__global__ void add_inplace(float* x, const float* h, long n) {
    long i = (long)blockIdx.x * blockDim.x + threadIdx.x;
    if (i < n) x[i] += h[i];
}

// ---------------------------------------------------------------------------
// pointnet: per-pillar feature build + 1x1 conv (9 -> 64)  => x1 [P,64,10]
// ---------------------------------------------------------------------------
__global__ void pfn_kernel(const float* __restrict__ voxels, const int* __restrict__ npv,
                           const int* __restrict__ coords, const float* __restrict__ pw,
                           float* __restrict__ x1)
{
    int p = blockIdx.x;
    __shared__ float feats[NPTSC][9];
    __shared__ float mean3[3];
    const float* vx = voxels + (long)p * NPTSC * 4;
    int np = npv[p];
    if (threadIdx.x < 3) {
        float s = 0.f;
        for (int t = 0; t < NPTSC; ++t) s += vx[t * 4 + threadIdx.x];
        mean3[threadIdx.x] = s / (float)np;
    }
    __syncthreads();
    float cx = (float)coords[p * 4 + 0] * VXC + XOFFC;
    float cy = (float)coords[p * 4 + 1] * VYC + YOFFC;
    for (int idx = threadIdx.x; idx < NPTSC * 9; idx += blockDim.x) {
        int t = idx / 9, c = idx - t * 9;     // compile-time divisors
        float v;
        if (t >= np)      v = 0.f;
        else if (c < 4)   v = vx[t * 4 + c];
        else if (c < 7)   v = vx[t * 4 + (c - 4)] - mean3[c - 4];
        else if (c == 7)  v = vx[t * 4 + 0] - cx;
        else              v = vx[t * 4 + 1] - cy;
        feats[t][c] = v;
    }
    __syncthreads();
    int o = threadIdx.x;   // 64 threads = 64 output channels
    const float* wrow = pw + o * 9;
    for (int t = 0; t < NPTSC; ++t) {
        float s = 0.f;
        for (int c = 0; c < 9; ++c) s += feats[t][c] * wrow[c];
        x1[((long)p * 64 + o) * NPTSC + t] = s;
    }
}

// ---------------------------------------------------------------------------
// global batchnorm stats over axes (P, L) per channel; layout [P, C, L]
// division-free: threads stride over P, inner loop walks L contiguously
// ---------------------------------------------------------------------------
__global__ void bn_stats(const float* __restrict__ x, float* __restrict__ mean,
                         float* __restrict__ var, long Pn, int C, int L)
{
    int c = blockIdx.x;
    __shared__ float ss[256], sq[256];
    float s = 0.f, q = 0.f;
    for (long p = threadIdx.x; p < Pn; p += blockDim.x) {
        const float* xp = x + (p * C + c) * (long)L;
        for (int l = 0; l < L; ++l) { float v = xp[l]; s += v; q += v * v; }
    }
    ss[threadIdx.x] = s; sq[threadIdx.x] = q;
    __syncthreads();
    for (int st = 128; st > 0; st >>= 1) {
        if (threadIdx.x < st) { ss[threadIdx.x] += ss[threadIdx.x + st]; sq[threadIdx.x] += sq[threadIdx.x + st]; }
        __syncthreads();
    }
    if (threadIdx.x == 0) {
        long tot = Pn * L;
        float m = ss[0] / (float)tot;
        mean[c] = m;
        var[c]  = sq[0] / (float)tot - m * m;
    }
}

// grid = (nblk, C); division-free
__global__ void bn_apply_relu(float* __restrict__ x, const float* __restrict__ mean,
                              const float* __restrict__ var, const float* __restrict__ g,
                              const float* __restrict__ b, long Pn, int C, int L)
{
    int c = blockIdx.y;
    float sc = rsqrtf(var[c] + EPSN) * g[c];
    float bb = b[c] - mean[c] * sc;
    long stride = (long)gridDim.x * blockDim.x;
    for (long p = (long)blockIdx.x * blockDim.x + threadIdx.x; p < Pn; p += stride) {
        float* xp = x + (p * C + c) * (long)L;
        for (int l = 0; l < L; ++l) {
            float v = xp[l] * sc + bb;
            xp[l] = v > 0.f ? v : 0.f;
        }
    }
}

// ---------------------------------------------------------------------------
// conv_pfn: x viewed as [P, 10, 64] (points-as-channels), conv1d k=3 pad=1
// ---------------------------------------------------------------------------
__global__ void cpfn_conv1(const float* __restrict__ x1, const float* __restrict__ w1,
                           float* __restrict__ y)  // y [P,10,64]
{
    long i = (long)blockIdx.x * blockDim.x + threadIdx.x;
    long total = (long)PNUM * 10 * 64;
    if (i >= total) return;
    int l = (int)(i & 63);
    long t = i >> 6;
    int co = (int)(t % 10);    // compile-time divisor
    long p = t / 10;
    float s = 0.f;
    for (int ci = 0; ci < 10; ++ci)
        for (int k = 0; k < 3; ++k) {
            int ll = l + k - 1;
            if (ll < 0 || ll >= 64) continue;
            // transposed view [p, point=ci, chan=ll] == x1[p, ll, ci]
            s += x1[(p * 64 + ll) * NPTSC + ci] * w1[(co * 10 + ci) * 3 + k];
        }
    y[i] = s;
}

__global__ void cpfn_conv2(const float* __restrict__ y, const float* __restrict__ w2,
                           float* __restrict__ vf)  // vf [P,64]
{
    long i = (long)blockIdx.x * blockDim.x + threadIdx.x;
    long total = (long)PNUM * 64;
    if (i >= total) return;
    int l = (int)(i & 63);
    long p = i >> 6;
    float s = 0.f;
    for (int ci = 0; ci < 10; ++ci)
        for (int k = 0; k < 3; ++k) {
            int ll = l + k - 1;
            if (ll < 0 || ll >= 64) continue;
            s += y[(p * 10 + ci) * 64 + ll] * w2[ci * 3 + k];
        }
    vf[i] = s;
}

// ---------------------------------------------------------------------------
// scatter pillar features into NCHW canvas [B,64,NX,NY]
// ---------------------------------------------------------------------------
__global__ void scatter_kernel(const float* __restrict__ vf, const int* __restrict__ coords,
                               float* __restrict__ canvas)
{
    long i = (long)blockIdx.x * blockDim.x + threadIdx.x;
    long total = (long)PNUM * 64;
    if (i >= total) return;
    int c = (int)(i & 63);
    long p = i >> 6;
    int xx = coords[p * 4 + 1];
    int yy = coords[p * 4 + 2];
    int b  = coords[p * 4 + 3];
    canvas[(((long)b * 64 + c) * NXC + xx) * NYC + yy] = vf[i];
}

// ---------------------------------------------------------------------------
// 3x3 conv, pad=1, implicit GEMM via WMMA f16->f32.
// 128 threads = 4 waves; the 4 waves SHARE one 16x64 im2col A-tile in LDS
// and each owns a 16-wide Cout slice (blockIdx.y covers Cout in tiles of 64).
// B fragments are contiguous in the row-major weight matrix -> loaded
// directly from global (no LDS). K unrolled by 64 => 2 wmma per iteration.
// Requires: Cout % 64 == 0 and (Cin*9) % 64 == 0 (holds for Cin=64/128/256).
// ---------------------------------------------------------------------------
__global__ void __launch_bounds__(128)
conv3x3_wmma(const float* __restrict__ x, const float* __restrict__ w,
             const float* __restrict__ bias, float* __restrict__ y,
             int Bn, int Cin, int H, int W, int Cout, int Ho, int Wo, int stride)
{
    __shared__ _Float16 As[2][16 * 32];
    __shared__ int  s_iy[16], s_ix[16];
    __shared__ long s_in[16];    // input batch base offset
    __shared__ long s_out[16];   // b*Cout*HoWo + pix, or -1 if out of range
    const int tid  = threadIdx.x;
    const int lane = tid & 31;
    const int wv   = tid >> 5;
    const long HoWo = (long)Ho * Wo;
    const long Mtot = (long)Bn * HoWo;
    const int  Ktot = Cin * 9;
    const int  n0   = blockIdx.y * 64 + wv * 16;

    // decode the 16 tile pixels ONCE (all runtime divisions live here)
    if (tid < 16) {
        long gm = (long)blockIdx.x * 16 + tid;
        if (gm < Mtot) {
            int b   = (int)(gm / HoWo);
            int pix = (int)(gm - (long)b * HoWo);
            int oy = pix / Wo, ox = pix - oy * Wo;
            s_iy[tid]  = oy * stride - 1;
            s_ix[tid]  = ox * stride - 1;
            s_in[tid]  = (long)b * Cin * H * W;
            s_out[tid] = (long)b * Cout * HoWo + pix;
        } else {
            s_iy[tid] = -0x40000000; s_ix[tid] = -0x40000000;
            s_in[tid] = 0; s_out[tid] = -1;
        }
    }
    __syncthreads();

    const int ml = lane & 15;
    const int hi = lane >> 4;
    // B fragment source: lane wants w[n0+ml][k0 + 8*hi + (e<8 ? e : e+8)]
    const float* wrow = w + (long)(n0 + ml) * Ktot + hi * 8;

    v8f acc = {};
    for (int k0 = 0; k0 < Ktot; k0 += 64) {
        __builtin_prefetch(wrow + k0 + 128, 0, 1);
        // stage two 16x32 A-chunks: 1024 halves, 8 per thread
        #pragma unroll
        for (int j = 0; j < 8; ++j) {
            int idx  = tid * 8 + j;          // 0..1023
            int half = idx >> 9;
            int r512 = idx & 511;
            int m  = r512 >> 5;
            int kk = r512 & 31;
            int k  = k0 + half * 32 + kk;
            int ci = k / 9;                  // compile-time divisor 9
            int rr = k - ci * 9;
            int ky = rr / 3;                 // compile-time divisor 3
            int kx = rr - ky * 3;
            int iy = s_iy[m] + ky;
            int ix = s_ix[m] + kx;
            float v = 0.f;
            if (iy >= 0 && iy < H && ix >= 0 && ix < W)
                v = x[s_in[m] + ((long)ci * H + iy) * W + ix];
            As[half][m * 32 + kk] = (_Float16)v;
        }
        __syncthreads();
        v16h af0{}, af1{}, bf0{}, bf1{};
        #pragma unroll
        for (int e = 0; e < 16; ++e) {
            int vv = e >> 1, o = e & 1;
            int ka = (vv < 4 ? (vv * 2) : (16 + (vv - 4) * 2)) + hi * 8 + o;
            af0[e] = As[0][ml * 32 + ka];
            af1[e] = As[1][ml * 32 + ka];
            bf0[e] = (_Float16)wrow[k0 + (e < 8 ? e : e + 8)];
            bf1[e] = (_Float16)wrow[k0 + 32 + (e < 8 ? e : e + 8)];
        }
        acc = __builtin_amdgcn_wmma_f32_16x16x32_f16(false, af0, false, bf0,
                                                     (short)0, acc, false, false);
        acc = __builtin_amdgcn_wmma_f32_16x16x32_f16(false, af1, false, bf1,
                                                     (short)0, acc, false, false);
        __syncthreads();
    }

    // writeback: C/D layout — lane holds N=lane&15, VGPR r holds M=r+8*hi
    const int oc = n0 + ml;
    float bv = bias[oc];
    #pragma unroll
    for (int r = 0; r < 8; ++r) {
        int m = r + hi * 8;
        long ob = s_out[m];
        if (ob >= 0)
            y[ob + (long)oc * HoWo] = acc[r] + bv;
    }
}

// ---------------------------------------------------------------------------
// InstanceNorm + ReLU: one block per (b,c) plane; optional channel remap into
// a wider output tensor (for the concat).
// ---------------------------------------------------------------------------
__global__ void inorm_relu(const float* __restrict__ x, float* __restrict__ y,
                           int C, int HW, int outCtot, int outCoff)
{
    long plane = blockIdx.x;
    int b = (int)(plane / C), c = (int)(plane % C);
    const float* xp = x + plane * HW;
    __shared__ float ss[256], sq[256];
    float s = 0.f, q = 0.f;
    for (int i = threadIdx.x; i < HW; i += blockDim.x) {
        float v = xp[i]; s += v; q += v * v;
    }
    ss[threadIdx.x] = s; sq[threadIdx.x] = q;
    __syncthreads();
    for (int st = 128; st > 0; st >>= 1) {
        if (threadIdx.x < st) { ss[threadIdx.x] += ss[threadIdx.x + st]; sq[threadIdx.x] += sq[threadIdx.x + st]; }
        __syncthreads();
    }
    float m   = ss[0] / (float)HW;
    float var = sq[0] / (float)HW - m * m;
    float inv = rsqrtf(var + EPSN);
    float* yp = y + ((long)b * outCtot + outCoff + c) * HW;
    for (int i = threadIdx.x; i < HW; i += blockDim.x) {
        float v = (xp[i] - m) * inv;
        yp[i] = v > 0.f ? v : 0.f;
    }
}

// ---------------------------------------------------------------------------
// ConvTranspose2d with kernel==stride (pixel-shuffle form); w is [Cin,Cout,s,s]
// ---------------------------------------------------------------------------
__global__ void deconv_kernel(const float* __restrict__ x, const float* __restrict__ w,
                              const float* __restrict__ bias, float* __restrict__ y,
                              int Bn, int Cin, int Hin, int Win, int Cout, int s)
{
    int Ho = Hin * s, Wo = Win * s;
    long total = (long)Bn * Cout * Ho * Wo;
    long i = (long)blockIdx.x * blockDim.x + threadIdx.x;
    if (i >= total) return;
    int ow = (int)(i % Wo); long t = i / Wo;
    int oh = (int)(t % Ho); t /= Ho;
    int o  = (int)(t % Cout);
    int n  = (int)(t / Cout);
    int h = oh / s, a  = oh - h * s;
    int wd = ow / s, bb = ow - wd * s;
    float acc = bias[o];
    const float* xp = x + ((long)n * Cin) * Hin * Win + (long)h * Win + wd;
    for (int c = 0; c < Cin; ++c)
        acc += xp[(long)c * Hin * Win] * w[(((long)c * Cout + o) * s + a) * s + bb];
    y[i] = acc;
}

// ---------------------------------------------------------------------------
// detection head: 1x1 conv over [B,320,H1,W1]; grid.y = output channel
// (only compile-time divisions remain)
// ---------------------------------------------------------------------------
__global__ void head_kernel(const float* __restrict__ x, const float* __restrict__ w,
                            const float* __restrict__ bias, float* __restrict__ out,
                            int co, long sec_base, long batch_stride, long head_off)
{
    int o = blockIdx.y;
    long i = (long)blockIdx.x * blockDim.x + threadIdx.x;  // over BATCH*HWF
    if (i >= (long)BATCH * HWF) return;
    int n   = (int)(i / HWF);       // compile-time divisor
    int pix = (int)(i - (long)n * HWF);
    const float* xp = x + (long)n * 320 * HWF + pix;
    const float* wp = w + (long)o * 320;
    float s = bias[o];
    for (int c = 0; c < 320; ++c) s += xp[(long)c * HWF] * wp[c];
    out[sec_base + (long)n * batch_stride + head_off + (long)pix * co + o] = s;
}

// ---------------------------------------------------------------------------
// host-side helpers
// ---------------------------------------------------------------------------
static void launch_conv(const float* x, const float* w, const float* b, float* y,
                        int Bn, int Cin, int H, int W, int Cout, int s, hipStream_t st)
{
    int Ho = (H - 1) / s + 1;
    int Wo = (W - 1) / s + 1;
    long Mtot = (long)Bn * Ho * Wo;
    dim3 g((unsigned)((Mtot + 15) / 16), (unsigned)(Cout / 64));
    conv3x3_wmma<<<g, 128, 0, st>>>(x, w, b, y, Bn, Cin, H, W, Cout, Ho, Wo, s);
}

static void launch_inorm(const float* x, float* y, int C, int HW,
                         int outCtot, int outCoff, hipStream_t st)
{
    inorm_relu<<<BATCH * C, 256, 0, st>>>(x, y, C, HW, outCtot, outCoff);
}

static void launch_add(float* x, const float* h, long n, hipStream_t st)
{
    add_inplace<<<(unsigned)((n + 255) / 256), 256, 0, st>>>(x, h, n);
}

static void launch_bn(float* x, const float* mean, const float* var,
                      const float* g, const float* b, long Pn, int C, int L, hipStream_t st)
{
    dim3 grid(128, (unsigned)C);
    bn_apply_relu<<<grid, 256, 0, st>>>(x, mean, var, g, b, Pn, C, L);
}

extern "C" void kernel_launch(void* const* d_in, const int* in_sizes, int n_in,
                              void* d_out, int out_size, void* d_ws, size_t ws_size,
                              hipStream_t stream)
{
    (void)in_sizes; (void)n_in; (void)out_size; (void)ws_size;

    const float* voxels = (const float*)d_in[0];
    const int*   npv    = (const int*)d_in[1];
    const int*   coords = (const int*)d_in[2];
    // params in jax pytree (alphabetical dict key) order:
    // 3..12  block1, 13..24 block2, 25..36 block3,
    // 37 cpfn_b1, 38 cpfn_b2, 39 cpfn_g1, 40 cpfn_g2, 41 cpfn_w1, 42 cpfn_w2,
    // 43/44 deconv1 b/w, 45/46 deconv2 b/w, 47/48 deconv3 b/w,
    // 49..66 heads {cyc_box,cyc_cls,cyc_dir,ped_box,ped_cls,ped_dir,
    //               veh_box,veh_cls,veh_dir} x {b,w},
    // 67 pfn_b, 68 pfn_g, 69 pfn_w
    const float* PR[70];
    for (int i = 0; i < 70; ++i) PR[i] = (const float*)d_in[i];

    float* ws = (float*)d_ws;
    size_t off = 0;
    auto alloc = [&](size_t n) { float* p = ws + off; off += n; return p; };

    float* pn1 = alloc((size_t)PNUM * 64 * NPTSC);   // 30.72M
    float* pn2 = alloc((size_t)PNUM * 10 * 64);      // 30.72M
    float* canvas = pn1;                             // reused: 54.85M fits in pn1+pn2
    float* vf   = alloc((size_t)PNUM * 64);
    float* mean = alloc(256);
    float* var  = alloc(256);
    float* t0 = alloc((size_t)BATCH * 64 * 216 * 248);
    float* t1 = alloc((size_t)BATCH * 64 * HWF);
    float* t2 = alloc((size_t)BATCH * 64 * HWF);
    float* t3 = alloc((size_t)BATCH * 64 * HWF);
    float* u0 = alloc((size_t)BATCH * 128 * 54 * 62);
    float* u1 = alloc((size_t)BATCH * 128 * 54 * 62);
    float* u2 = alloc((size_t)BATCH * 128 * 54 * 62);
    float* v0 = alloc((size_t)BATCH * 256 * 27 * 31);
    float* v1 = alloc((size_t)BATCH * 256 * 27 * 31);
    float* v2 = alloc((size_t)BATCH * 256 * 27 * 31);
    float* dd1 = alloc((size_t)BATCH * 64 * HWF);
    float* dd2 = alloc((size_t)BATCH * 128 * HWF);
    float* dd3 = alloc((size_t)BATCH * 128 * HWF);
    float* cat = alloc((size_t)BATCH * 320 * HWF);

    float* dout = (float*)d_out;

    // ---- pointnet ----
    pfn_kernel<<<PNUM, 64, 0, stream>>>(voxels, npv, coords, PR[69], pn1);
    bn_stats<<<64, 256, 0, stream>>>(pn1, mean, var, PNUM, 64, NPTSC);
    launch_bn(pn1, mean, var, PR[68], PR[67], PNUM, 64, NPTSC, stream);
    cpfn_conv1<<<(unsigned)(((long)PNUM * 640 + 255) / 256), 256, 0, stream>>>(pn1, PR[41], pn2);
    bn_stats<<<10, 256, 0, stream>>>(pn2, mean, var, PNUM, 10, 64);
    launch_bn(pn2, mean, var, PR[39], PR[37], PNUM, 10, 64, stream);
    cpfn_conv2<<<(unsigned)(((long)PNUM * 64 + 255) / 256), 256, 0, stream>>>(pn2, PR[42], vf);
    bn_stats<<<1, 256, 0, stream>>>(vf, mean, var, PNUM, 1, 64);
    launch_bn(vf, mean, var, PR[40], PR[38], PNUM, 1, 64, stream);

    // ---- scatter ----
    {
        long n = (long)BATCH * 64 * NXC * NYC;
        zero_f32<<<(unsigned)((n + 255) / 256), 256, 0, stream>>>(canvas, n);
        scatter_kernel<<<(unsigned)(((long)PNUM * 64 + 255) / 256), 256, 0, stream>>>(vf, coords, canvas);
    }

    // ---- block1 ----
    launch_conv(canvas, PR[4], PR[3], t0, BATCH, 64, NXC, NYC, 64, 2, stream);
    launch_inorm(t0, t0, 64, 216 * 248, 64, 0, stream);
    launch_conv(t0, PR[6], PR[5], t1, BATCH, 64, 216, 248, 64, 2, stream);
    launch_inorm(t1, t1, 64, HWF, 64, 0, stream);
    long n64 = (long)BATCH * 64 * HWF;
    // res group 0 (2 convs)
    launch_inorm(t1, t2, 64, HWF, 64, 0, stream);
    launch_conv(t2, PR[8], PR[7], t3, BATCH, 64, H1, W1, 64, 1, stream);
    launch_inorm(t3, t3, 64, HWF, 64, 0, stream);
    launch_conv(t3, PR[10], PR[9], t2, BATCH, 64, H1, W1, 64, 1, stream);
    launch_add(t1, t2, n64, stream);
    // res group 1 (1 conv)
    launch_inorm(t1, t2, 64, HWF, 64, 0, stream);
    launch_conv(t2, PR[12], PR[11], t3, BATCH, 64, H1, W1, 64, 1, stream);
    launch_add(t1, t3, n64, stream);

    // up1 = relu(inorm(deconv1(t1))), into cat channels [0,64)
    {
        long n = (long)BATCH * 64 * HWF;
        deconv_kernel<<<(unsigned)((n + 255) / 256), 256, 0, stream>>>(
            t1, PR[44], PR[43], dd1, BATCH, 64, H1, W1, 64, 1);
        launch_inorm(dd1, cat, 64, HWF, 320, 0, stream);
    }

    // ---- block2 ----
    launch_conv(t1, PR[14], PR[13], u0, BATCH, 64, H1, W1, 128, 2, stream);
    launch_inorm(u0, u0, 128, 54 * 62, 128, 0, stream);
    long n128 = (long)BATCH * 128 * 54 * 62;
    // res 0 (2 convs)
    launch_inorm(u0, u1, 128, 54 * 62, 128, 0, stream);
    launch_conv(u1, PR[16], PR[15], u2, BATCH, 128, 54, 62, 128, 1, stream);
    launch_inorm(u2, u2, 128, 54 * 62, 128, 0, stream);
    launch_conv(u2, PR[18], PR[17], u1, BATCH, 128, 54, 62, 128, 1, stream);
    launch_add(u0, u1, n128, stream);
    // res 1 (2 convs)
    launch_inorm(u0, u1, 128, 54 * 62, 128, 0, stream);
    launch_conv(u1, PR[20], PR[19], u2, BATCH, 128, 54, 62, 128, 1, stream);
    launch_inorm(u2, u2, 128, 54 * 62, 128, 0, stream);
    launch_conv(u2, PR[22], PR[21], u1, BATCH, 128, 54, 62, 128, 1, stream);
    launch_add(u0, u1, n128, stream);
    // res 2 (1 conv)
    launch_inorm(u0, u1, 128, 54 * 62, 128, 0, stream);
    launch_conv(u1, PR[24], PR[23], u2, BATCH, 128, 54, 62, 128, 1, stream);
    launch_add(u0, u2, n128, stream);

    // up2 into cat channels [64,192)
    {
        long n = (long)BATCH * 128 * HWF;
        deconv_kernel<<<(unsigned)((n + 255) / 256), 256, 0, stream>>>(
            u0, PR[46], PR[45], dd2, BATCH, 128, 54, 62, 128, 2);
        launch_inorm(dd2, cat, 128, HWF, 320, 64, stream);
    }

    // ---- block3 ----
    launch_conv(u0, PR[26], PR[25], v0, BATCH, 128, 54, 62, 256, 2, stream);
    launch_inorm(v0, v0, 256, 27 * 31, 256, 0, stream);
    long n256 = (long)BATCH * 256 * 27 * 31;
    // res 0 (2 convs)
    launch_inorm(v0, v1, 256, 27 * 31, 256, 0, stream);
    launch_conv(v1, PR[28], PR[27], v2, BATCH, 256, 27, 31, 256, 1, stream);
    launch_inorm(v2, v2, 256, 27 * 31, 256, 0, stream);
    launch_conv(v2, PR[30], PR[29], v1, BATCH, 256, 27, 31, 256, 1, stream);
    launch_add(v0, v1, n256, stream);
    // res 1 (2 convs)
    launch_inorm(v0, v1, 256, 27 * 31, 256, 0, stream);
    launch_conv(v1, PR[32], PR[31], v2, BATCH, 256, 27, 31, 256, 1, stream);
    launch_inorm(v2, v2, 256, 27 * 31, 256, 0, stream);
    launch_conv(v2, PR[34], PR[33], v1, BATCH, 256, 27, 31, 256, 1, stream);
    launch_add(v0, v1, n256, stream);
    // res 2 (1 conv)
    launch_inorm(v0, v1, 256, 27 * 31, 256, 0, stream);
    launch_conv(v1, PR[36], PR[35], v2, BATCH, 256, 27, 31, 256, 1, stream);
    launch_add(v0, v2, n256, stream);

    // up3 into cat channels [192,320)
    {
        long n = (long)BATCH * 128 * HWF;
        deconv_kernel<<<(unsigned)((n + 255) / 256), 256, 0, stream>>>(
            v0, PR[48], PR[47], dd3, BATCH, 256, 27, 31, 128, 4);
        launch_inorm(dd3, cat, 128, HWF, 320, 192, stream);
    }

    // ---- heads ----
    long cls_base = 0, cls_bs = (long)HWF * 9;                       // [B, HW*9, 1]
    long box_base = (long)BATCH * HWF * 9, box_bs = (long)HWF * 63;  // [B, HW*9, 7]
    long dir_base = box_base + (long)BATCH * HWF * 63, dir_bs = (long)HWF * 18; // [B, HW*9, 2]
    unsigned hgrid = (unsigned)(((long)BATCH * HWF + 255) / 256);
    auto head = [&](const float* w, const float* b, int co, long base, long bs, long hoff) {
        dim3 g(hgrid, (unsigned)co);
        head_kernel<<<g, 256, 0, stream>>>(cat, w, b, dout, co, base, bs, hoff);
    };
    head(PR[64], PR[63], 6,  cls_base, cls_bs, 0);                  // veh_cls
    head(PR[58], PR[57], 1,  cls_base, cls_bs, (long)HWF * 6);      // ped_cls
    head(PR[52], PR[51], 2,  cls_base, cls_bs, (long)HWF * 7);      // cyc_cls
    head(PR[62], PR[61], 42, box_base, box_bs, 0);                  // veh_box
    head(PR[56], PR[55], 7,  box_base, box_bs, (long)HWF * 42);     // ped_box
    head(PR[50], PR[49], 14, box_base, box_bs, (long)HWF * 49);     // cyc_box
    head(PR[66], PR[65], 12, dir_base, dir_bs, 0);                  // veh_dir
    head(PR[60], PR[59], 2,  dir_base, dir_bs, (long)HWF * 12);     // ped_dir
    head(PR[54], PR[53], 4,  dir_base, dir_bs, (long)HWF * 14);     // cyc_dir
}
